// TimeFeatureBlock_14998025797904
// MI455X (gfx1250) — compile-verified, matrix operand order checked by reading, not run on hardware
//
#include <hip/hip_runtime.h>
#include <hip/hip_bf16.h>
#include <math.h>

// ---------------------------------------------------------------------------
// CDNA5 (gfx1250) wave32 WMMA types / helpers
// ---------------------------------------------------------------------------
typedef __attribute__((ext_vector_type(16))) _Float16 v16h;
typedef __attribute__((ext_vector_type(8)))  _Float16 v8h;
typedef __attribute__((ext_vector_type(8)))  float    v8f;
typedef __attribute__((ext_vector_type(4)))  unsigned int u32x4;
typedef __attribute__((ext_vector_type(8)))  int i32x8;
typedef __attribute__((ext_vector_type(4)))  int i32x4;

union Frag16 { v16h v; v8h h[2]; };

// ISA 16-bit A/B fragment (16x32, wave32): lane half = lane>>4 reads two
// contiguous 8-element K-runs: [half*8, +8) and [16+half*8, +8).
__device__ __forceinline__ v16h load_frag(const _Float16* rowbase, int half) {
  Frag16 f;
  f.h[0] = *reinterpret_cast<const v8h*>(rowbase + half * 8);
  f.h[1] = *reinterpret_cast<const v8h*>(rowbase + 16 + half * 8);
  return f.v;
}

__device__ __forceinline__ v8f wmma_f16(v16h a, v16h b, v8f c) {
  return __builtin_amdgcn_wmma_f32_16x16x32_f16(
      false, a, false, b, (short)0, c, false, false);
}

__device__ __forceinline__ float gelu_exact(float x) {
  return 0.5f * x * (1.0f + erff(x * 0.70710678118654752f));
}
__device__ __forceinline__ float sigmoidf_(float x) {
  return 1.0f / (1.0f + __expf(-x));
}

// ---------------------------------------------------------------------------
// Tensor Data Mover: one instruction moves a (t1 x 32) f16 tile (row-major,
// 4-dword LDS pad every 16 dwords -> 40-f16 LDS pitch) global -> LDS.
// ---------------------------------------------------------------------------
#if defined(__has_builtin)
#  if __has_builtin(__builtin_amdgcn_tensor_load_to_lds) && \
      __has_builtin(__builtin_amdgcn_s_wait_tensorcnt)
#    define USE_TDM 1
#  endif
#endif

#ifndef USE_TDM
#warning "gfx1250 TDM builtins unavailable; falling back to manual LDS copies"
#endif

#ifdef USE_TDM
__device__ __forceinline__ void tdm_load_tile(const _Float16* g, unsigned lds_byte_off,
                                              unsigned d0 /*row len (elems)*/,
                                              unsigned d1 /*rows*/,
                                              unsigned t0 /*tile cols*/,
                                              unsigned t1 /*tile rows*/)
{
  unsigned long long ga = (unsigned long long)(size_t)g;
  u32x4 g0 = { 1u,                                   // count=1, gather off
               lds_byte_off,                         // lds_addr
               (unsigned)ga,                         // global_addr[31:0]
               (unsigned)((ga >> 32) & 0x1FFFFFFu) | (2u << 30) };  // type=2
  // data_size=2B (code 1), pad_enable, pad_interval=16dw (3), pad_amount=4dw (3)
  i32x8 g1 = { (int)((1u << 16) | (1u << 20) | (3u << 22) | (3u << 25)),
               (int)((d0 & 0xFFFFu) << 16),
               (int)(((d0 >> 16) & 0xFFFFu) | ((d1 & 0xFFFFu) << 16)),
               (int)(((d1 >> 16) & 0xFFFFu) | ((t0 & 0xFFFFu) << 16)),
               (int)(t1 & 0xFFFFu),
               (int)d0,
               0, 0 };
  i32x4 gz = { 0, 0, 0, 0 };
#if defined(__clang_major__) && __clang_major__ >= 23
  i32x8 gz8 = { 0, 0, 0, 0, 0, 0, 0, 0 };
  __builtin_amdgcn_tensor_load_to_lds(g0, g1, gz, gz, gz8, 0);
#else
  __builtin_amdgcn_tensor_load_to_lds(g0, g1, gz, gz, 0);
#endif
}
#endif

// Model dims
static constexpr int kB   = 256;
static constexpr int kL   = 128;
static constexpr int kCIN = 75;
static constexpr int kDE  = 128;
static constexpr int kDM  = 126;
static constexpr int kNH  = 6;
static constexpr int kDH  = 21;
static constexpr int kDFF = 2048;
static constexpr int kQKV = 3 * kDM;  // fused q|k|v width = 378
static constexpr int LDSB = 40;       // padded LDS pitch in f16 (80B)

// ---------------------------------------------------------------------------
// 0a. Weight repack: f32 (K x N) -> f16 transposed (N x K)
// ---------------------------------------------------------------------------
__global__ __launch_bounds__(256)
void transpose_f16_kernel(const float* __restrict__ src, _Float16* __restrict__ dst,
                          int K, int N)
{
  __shared__ float tile[32][33];
  int k0 = blockIdx.x * 32, n0 = blockIdx.y * 32;
  int tx = threadIdx.x & 31, ty = threadIdx.x >> 5;
  #pragma unroll
  for (int it = 0; it < 4; ++it) {
    int r = ty + it * 8;
    int gk = k0 + r, gn = n0 + tx;
    tile[r][tx] = (gk < K && gn < N) ? src[(size_t)gk * N + gn] : 0.f;
  }
  __syncthreads();
  #pragma unroll
  for (int it = 0; it < 4; ++it) {
    int r = ty + it * 8;
    int gn = n0 + r, gk = k0 + tx;
    if (gn < N && gk < K) dst[(size_t)gn * K + gk] = (_Float16)tile[tx][r];
  }
}

// 0b. Fused q|k|v bias concat (378 per layer)
__global__ __launch_bounds__(128)
void qkv_bias_kernel(const float* __restrict__ qb, const float* __restrict__ kb,
                     const float* __restrict__ vb, float* __restrict__ dst)
{
  int li = blockIdx.x, t = threadIdx.x;
  if (t < kDM) {
    dst[(size_t)li * kQKV + t]           = qb[(size_t)li * kDM + t];
    dst[(size_t)li * kQKV + kDM + t]     = kb[(size_t)li * kDM + t];
    dst[(size_t)li * kQKV + 2 * kDM + t] = vb[(size_t)li * kDM + t];
  }
}

// ---------------------------------------------------------------------------
// 1. Fused embedding: wrap-conv1d token embed + sinusoidal pos + time linear
// ---------------------------------------------------------------------------
__global__ __launch_bounds__(128)
void embed_kernel(const float* __restrict__ x_enc, const float* __restrict__ x_mark,
                  const float* __restrict__ token_w, const float* __restrict__ time_w,
                  const float* __restrict__ time_b, float* __restrict__ emb)
{
  int b = blockIdx.x, l = blockIdx.y, c = threadIdx.x;
  __shared__ float rows[3 * kCIN];
  __shared__ float mark[3];
  #pragma unroll
  for (int it = 0; it < 2; ++it) {
    int e = threadIdx.x + it * 128;
    if (e < 3 * kCIN) {
      int kk = e / kCIN, i = e - kk * kCIN;
      int ll = (l + kk - 1 + kL) & (kL - 1);
      rows[e] = x_enc[((size_t)b * kL + ll) * kCIN + i];
    }
  }
  if (threadIdx.x < 3) mark[threadIdx.x] = x_mark[((size_t)b * kL + l) * 3 + threadIdx.x];
  __syncthreads();

  float s = 0.f;
  const float* w = token_w + (size_t)c * kCIN * 3;
  #pragma unroll 5
  for (int i = 0; i < kCIN; ++i) {
    s += w[i * 3 + 0] * rows[0 * kCIN + i];
    s += w[i * 3 + 1] * rows[1 * kCIN + i];
    s += w[i * 3 + 2] * rows[2 * kCIN + i];
  }
  int cc = c & ~1;
  float div = __expf((float)cc * (-9.210340371976184f / (float)kDE));
  float ang = (float)l * div;
  float pe  = (c & 1) ? __cosf(ang) : __sinf(ang);
  float t = time_b[c];
  for (int m = 0; m < 3; ++m) t += mark[m] * time_w[m * kDE + c];
  emb[((size_t)b * kL + l) * kDE + c] = s + pe + t;
}

// 2. GAP over length
__global__ __launch_bounds__(128)
void gap_kernel(const float* __restrict__ emb, float* __restrict__ gap)
{
  int b = blockIdx.x, c = threadIdx.x;
  float s = 0.f;
  for (int l = 0; l < kL; ++l) s += emb[((size_t)b * kL + l) * kDE + c];
  gap[(size_t)b * kDE + c] = s * (1.0f / (float)kL);
}

// 3. ODConv attention vectors (ch 128 | fil 128 | sp 3 | ker 4, stride 264)
__global__ __launch_bounds__(128)
void odatt_kernel(const float* __restrict__ gap,
                  const float* __restrict__ fc_w, const float* __restrict__ bn_g,
                  const float* __restrict__ bn_b, const float* __restrict__ ch_w,
                  const float* __restrict__ ch_b, const float* __restrict__ fil_w,
                  const float* __restrict__ fil_b, const float* __restrict__ sp_w,
                  const float* __restrict__ sp_b, const float* __restrict__ ker_w,
                  const float* __restrict__ ker_b, float* __restrict__ attv)
{
  int b = blockIdx.x, c = threadIdx.x;
  __shared__ float gsh[kDE];
  __shared__ float ash[16];
  __shared__ float ksh[4];
  gsh[c] = gap[(size_t)b * kDE + c];
  __syncthreads();
  if (c < 16) {
    float s = 0.f;
    for (int i = 0; i < kDE; ++i) s += gsh[i] * fc_w[i * 16 + c];
    s = s * (bn_g[c] * rsqrtf(1.0f + 1e-5f)) + bn_b[c];
    ash[c] = fmaxf(s, 0.f);
  }
  __syncthreads();
  float sc = ch_b[c], sf = fil_b[c];
  #pragma unroll
  for (int n = 0; n < 16; ++n) {
    sc += ash[n] * ch_w[n * kDE + c];
    sf += ash[n] * fil_w[n * kDE + c];
  }
  float* av = attv + (size_t)b * 264;
  av[c]       = sigmoidf_(sc);
  av[128 + c] = sigmoidf_(sf);
  if (c < 3) {
    float s = sp_b[c];
    for (int n = 0; n < 16; ++n) s += ash[n] * sp_w[n * 3 + c];
    av[256 + c] = sigmoidf_(s);
  }
  if (c < 4) {
    float s = ker_b[c];
    for (int n = 0; n < 16; ++n) s += ash[n] * ker_w[n * 4 + c];
    ksh[c] = s;
  }
  __syncthreads();
  if (c < 4) {
    float mx = fmaxf(fmaxf(ksh[0], ksh[1]), fmaxf(ksh[2], ksh[3]));
    float sum = __expf(ksh[0]-mx) + __expf(ksh[1]-mx) + __expf(ksh[2]-mx) + __expf(ksh[3]-mx);
    av[260 + c] = __expf(ksh[c] - mx) / sum;
  }
}

// 4a. xc = emb * ch_att, f16, layout (B, L, 128)
__global__ __launch_bounds__(128)
void xch_kernel(const float* __restrict__ emb, const float* __restrict__ attv,
                _Float16* __restrict__ xch)
{
  int b = blockIdx.x, l = blockIdx.y, i = threadIdx.x;
  float ch = attv[(size_t)b * 264 + i];
  size_t idx = ((size_t)b * kL + l) * kDE + i;
  xch[idx] = (_Float16)(emb[idx] * ch);
}

// 4b. Aggregate dynamic-conv weight, f16, REORDERED K: k = j*128 + i
__global__ __launch_bounds__(128)
void aggw_kernel(const float* __restrict__ od_weight, const float* __restrict__ attv,
                 _Float16* __restrict__ aggh)
{
  int b = blockIdx.x, o = blockIdx.y, i = threadIdx.x;
  const float* av = attv + (size_t)b * 264;
  float ker0 = av[260], ker1 = av[261], ker2 = av[262], ker3 = av[263];
  _Float16* dst = aggh + ((size_t)b * kDE + o) * 384;
  #pragma unroll
  for (int j = 0; j < 3; ++j) {
    size_t idx = (size_t)(o * kDE + i) * 3 + j;
    size_t str = (size_t)kDE * kDE * 3;
    float s = ker0 * od_weight[idx] + ker1 * od_weight[idx + str] +
              ker2 * od_weight[idx + 2*str] + ker3 * od_weight[idx + 3*str];
    dst[j * kDE + i] = (_Float16)(s * av[256 + j]);   // k = j*128 + i
  }
}

// ---------------------------------------------------------------------------
// 5. Dynamic conv: batched WMMA GEMM; A and B tiles via DOUBLE-BUFFERED TDM.
//    With k = j*128+i, the B tile for fixed j is the contiguous 2D slab
//    xc[j+t][i].  Block = 64(o) x 128(t); wave = 32x32 C (4 WMMA / 4 frags).
//    grid(2, B), block(256)
// ---------------------------------------------------------------------------
__global__ __launch_bounds__(256)
void dynconv_kernel(const _Float16* __restrict__ aggh, const _Float16* __restrict__ xch,
                    const float* __restrict__ attv, float* __restrict__ out,
                    _Float16* __restrict__ outh)
{
  __shared__ _Float16 As[2][64 * LDSB];
  __shared__ _Float16 Bs[2][128 * LDSB];
  int mblk = blockIdx.x, b = blockIdx.y;
  int m0 = mblk * 64;
  const _Float16* A  = aggh + (size_t)b * kDE * 384;
  const _Float16* Eh = xch + (size_t)b * kL * kDE;
  const float* fil   = attv + (size_t)b * 264 + 128;
  int tid = threadIdx.x, lane = tid & 31, wave = tid >> 5;
  int m2 = wave & 1, n4 = wave >> 1;
  int half = lane >> 4, lc = lane & 15;
  v8f acc[2][2] = {};
  // 12 K-steps: idx -> j = idx>>2, c0 = (idx&3)*32
  auto aAddr = [&](int idx) { return A  + (size_t)m0 * 384 + (idx >> 2) * kDE + (idx & 3) * 32; };
  auto bAddr = [&](int idx) { return Eh + (size_t)(idx >> 2) * kDE + (idx & 3) * 32; };
#ifdef USE_TDM
  if (wave == 0) {
    tdm_load_tile(aAddr(0), (unsigned)(size_t)(void*)As[0], 384, 128, 32, 64);
    tdm_load_tile(bAddr(0), (unsigned)(size_t)(void*)Bs[0], kDE, kL, 32, 128);
  }
#endif
  int buf = 0;
  for (int idx = 0; idx < 12; ++idx, buf ^= 1) {
#ifdef USE_TDM
    if (wave == 0) {
      if (idx + 1 < 12) {
        tdm_load_tile(aAddr(idx + 1), (unsigned)(size_t)(void*)As[buf ^ 1], 384, 128, 32, 64);
        tdm_load_tile(bAddr(idx + 1), (unsigned)(size_t)(void*)Bs[buf ^ 1], kDE, kL, 32, 128);
        __builtin_amdgcn_s_wait_tensorcnt(2);   // current pair has landed
      } else {
        __builtin_amdgcn_s_wait_tensorcnt(0);
      }
    }
#else
    {
      const _Float16* Ai = aAddr(idx);
      const _Float16* Bi = bAddr(idx);
      int r = tid >> 2, cb = (tid & 3) * 8;
      #pragma unroll
      for (int u = 0; u < 8; ++u)
        As[buf][r * LDSB + cb + u] = Ai[(size_t)r * 384 + cb + u];
      int rn = tid >> 1, cn = (tid & 1) * 16;
      int row = (idx >> 2) + rn;
      #pragma unroll
      for (int u = 0; u < 16; ++u)
        Bs[buf][rn * LDSB + cn + u] =
            (row < kL) ? Bi[(size_t)rn * kDE + cn + u] : (_Float16)0.f;
    }
#endif
    __syncthreads();   // current buffer ready for all waves
    const _Float16* as = As[buf];
    const _Float16* bs = Bs[buf];
    v16h a0 = load_frag(&as[(m2 * 32 + lc) * LDSB], half);
    v16h a1 = load_frag(&as[(m2 * 32 + 16 + lc) * LDSB], half);
    v16h b0 = load_frag(&bs[(n4 * 32 + lc) * LDSB], half);
    v16h b1 = load_frag(&bs[(n4 * 32 + 16 + lc) * LDSB], half);
    acc[0][0] = wmma_f16(a0, b0, acc[0][0]);
    acc[0][1] = wmma_f16(a0, b1, acc[0][1]);
    acc[1][0] = wmma_f16(a1, b0, acc[1][0]);
    acc[1][1] = wmma_f16(a1, b1, acc[1][1]);
    __syncthreads();   // reads done before this buffer is refilled
  }
  #pragma unroll
  for (int mi = 0; mi < 2; ++mi) {
    #pragma unroll
    for (int r = 0; r < 8; ++r) {
      int o = m0 + m2 * 32 + mi * 16 + r + half * 8;
      float f = fil[o];
      size_t base = ((size_t)b * kDE + o) * kDM;
      #pragma unroll
      for (int ni = 0; ni < 2; ++ni) {
        int t = n4 * 32 + ni * 16 + lc;
        if (t < kDM) {
          float v = acc[mi][ni][r] * f;
          out[base + t] = v;
          outh[base + t] = (_Float16)v;
        }
      }
    }
  }
}

// ---------------------------------------------------------------------------
// 6. Generic WMMA GEMM.  A: f16 (M x K).  Wt: f16 (N x K) pre-transposed.
//    Block 64(M) x 128(N); wave = 32x32 C (4 WMMA / 4 frags / K-step).
//    DOUBLE-BUFFERED TDM overlaps the next tile DMA with current WMMAs.
// ---------------------------------------------------------------------------
__global__ __launch_bounds__(256)
void gemm_kernel(const _Float16* __restrict__ A, const _Float16* __restrict__ Wt,
                 const float* __restrict__ bias, float* __restrict__ C,
                 _Float16* __restrict__ Ch, int M, int N, int K, int act)
{
  __shared__ _Float16 As[2][64 * LDSB];
  __shared__ _Float16 Bs[2][128 * LDSB];
  int m0 = blockIdx.y * 64;
  int n0 = blockIdx.x * 128;
  int tid = threadIdx.x, lane = tid & 31, wave = tid >> 5;
  int m2 = wave & 1, n4 = wave >> 1;
  int half = lane >> 4, lc = lane & 15;
  v8f acc[2][2] = {};
#ifdef USE_TDM
  if (wave == 0) {
    tdm_load_tile(A  + (size_t)m0 * K, (unsigned)(size_t)(void*)As[0], K, M, 32, 64);
    tdm_load_tile(Wt + (size_t)n0 * K, (unsigned)(size_t)(void*)Bs[0], K, N, 32, 128);
  }
#endif
  int buf = 0;
  for (int k0 = 0; k0 < K; k0 += 32, buf ^= 1) {
#ifdef USE_TDM
    if (wave == 0) {
      if (k0 + 32 < K) {
        tdm_load_tile(A  + (size_t)m0 * K + k0 + 32,
                      (unsigned)(size_t)(void*)As[buf ^ 1], K, M, 32, 64);
        tdm_load_tile(Wt + (size_t)n0 * K + k0 + 32,
                      (unsigned)(size_t)(void*)Bs[buf ^ 1], K, N, 32, 128);
        __builtin_amdgcn_s_wait_tensorcnt(2);   // current pair has landed
      } else {
        __builtin_amdgcn_s_wait_tensorcnt(0);
      }
    }
#else
    {
      int r = tid >> 2, cb = (tid & 3) * 8;
      int gm = m0 + r;
      #pragma unroll
      for (int u = 0; u < 8; ++u) {
        int gk = k0 + cb + u;
        As[buf][r * LDSB + cb + u] =
            (gm < M && gk < K) ? A[(size_t)gm * K + gk] : (_Float16)0.f;
      }
      int rn = tid >> 1, cn = (tid & 1) * 16;
      int gn = n0 + rn;
      #pragma unroll
      for (int u = 0; u < 16; ++u) {
        int gk = k0 + cn + u;
        Bs[buf][rn * LDSB + cn + u] =
            (gn < N && gk < K) ? Wt[(size_t)gn * K + gk] : (_Float16)0.f;
      }
    }
#endif
    __syncthreads();
    const _Float16* as = As[buf];
    const _Float16* bs = Bs[buf];
    v16h a0 = load_frag(&as[(m2 * 32 + lc) * LDSB], half);
    v16h a1 = load_frag(&as[(m2 * 32 + 16 + lc) * LDSB], half);
    v16h b0 = load_frag(&bs[(n4 * 32 + lc) * LDSB], half);
    v16h b1 = load_frag(&bs[(n4 * 32 + 16 + lc) * LDSB], half);
    acc[0][0] = wmma_f16(a0, b0, acc[0][0]);
    acc[0][1] = wmma_f16(a0, b1, acc[0][1]);
    acc[1][0] = wmma_f16(a1, b0, acc[1][0]);
    acc[1][1] = wmma_f16(a1, b1, acc[1][1]);
    __syncthreads();
  }
  #pragma unroll
  for (int mi = 0; mi < 2; ++mi) {
    #pragma unroll
    for (int r = 0; r < 8; ++r) {
      int gm = m0 + m2 * 32 + mi * 16 + r + half * 8;
      if (gm >= M) continue;
      #pragma unroll
      for (int ni = 0; ni < 2; ++ni) {
        int gn = n0 + n4 * 32 + ni * 16 + lc;
        if (gn >= N) continue;
        float v = acc[mi][ni][r];
        if (bias) v += bias[gn];
        if (act == 1) v = gelu_exact(v);
        if (C)  C[(size_t)gm * N + gn] = v;
        if (Ch) Ch[(size_t)gm * N + gn] = (_Float16)v;
      }
    }
  }
}

// ---------------------------------------------------------------------------
// 7. Fused attention per (head, batch): softmax(Q Kt * scale) V in LDS.
//    Input is the fused q|k|v buffer (pitch 378), f16 ctx out (pitch 126).
// ---------------------------------------------------------------------------
template <int LQ>
__global__ __launch_bounds__(256)
void attention_kernel(const _Float16* __restrict__ QKV, _Float16* __restrict__ ctx,
                      float scale)
{
  __shared__ _Float16 Qs[128 * 40];
  __shared__ _Float16 Ks[128 * 40];
  __shared__ _Float16 Vt[32 * 136];
  __shared__ _Float16 Ps[128 * 136];
  constexpr int nT = LQ / 16;
  int h = blockIdx.x, b = blockIdx.y;
  int tid = threadIdx.x, lane = tid & 31, wave = tid >> 5;
  const _Float16* Qb = QKV + ((size_t)b * LQ) * kQKV + h * kDH;
  const _Float16* Kb = Qb + kDM;
  const _Float16* Vb = Qb + 2 * kDM;
  #pragma unroll
  for (int it = 0; it < (LQ * 32) / 256; ++it) {
    int e = tid + it * 256;
    int r = e >> 5, c = e & 31;
    _Float16 qv = (c < kDH) ? Qb[(size_t)r * kQKV + c] : (_Float16)0.f;
    _Float16 kv = (c < kDH) ? Kb[(size_t)r * kQKV + c] : (_Float16)0.f;
    _Float16 vv = (c < kDH) ? Vb[(size_t)r * kQKV + c] : (_Float16)0.f;
    Qs[r * 40 + c] = qv;
    Ks[r * 40 + c] = kv;
    Vt[c * 136 + r] = vv;
  }
  __syncthreads();
  int half = lane >> 4, lc = lane & 15;
  if (wave < nT) {
    v8f accS[nT];
    v16h a = load_frag(&Qs[(wave * 16 + lc) * 40], half);
    #pragma unroll
    for (int t = 0; t < nT; ++t) {
      v16h bb = load_frag(&Ks[(t * 16 + lc) * 40], half);
      v8f z = {};
      accS[t] = wmma_f16(a, bb, z);
    }
    #pragma unroll
    for (int r = 0; r < 8; ++r) {
      float mx = -3.0e38f;
      #pragma unroll
      for (int t = 0; t < nT; ++t) { accS[t][r] *= scale; mx = fmaxf(mx, accS[t][r]); }
      for (int m = 1; m < 16; m <<= 1) mx = fmaxf(mx, __shfl_xor(mx, m, 32));
      float sum = 0.f;
      #pragma unroll
      for (int t = 0; t < nT; ++t) { float e2 = __expf(accS[t][r] - mx); accS[t][r] = e2; sum += e2; }
      for (int m = 1; m < 16; m <<= 1) sum += __shfl_xor(sum, m, 32);
      float inv = 1.0f / sum;
      int row = wave * 16 + r + half * 8;
      #pragma unroll
      for (int t = 0; t < nT; ++t) Ps[row * 136 + lc + 16 * t] = (_Float16)(accS[t][r] * inv);
    }
  }
  __syncthreads();
  if (wave < nT) {
    v8f c0 = {}, c1 = {};
    #pragma unroll
    for (int k0 = 0; k0 < LQ; k0 += 32) {
      v16h a  = load_frag(&Ps[(wave * 16 + lc) * 136 + k0], half);
      v16h b0 = load_frag(&Vt[lc * 136 + k0], half);
      v16h b1 = load_frag(&Vt[(16 + lc) * 136 + k0], half);
      c0 = wmma_f16(a, b0, c0);
      c1 = wmma_f16(a, b1, c1);
    }
    #pragma unroll
    for (int r = 0; r < 8; ++r) {
      int row = wave * 16 + r + half * 8;
      size_t base = ((size_t)b * LQ + row) * kDM + (size_t)h * kDH;
      ctx[base + lc] = (_Float16)c0[r];
      if (lc < kDH - 16) ctx[base + 16 + lc] = (_Float16)c1[r];
    }
  }
}

// 8. Residual-add + LayerNorm over D=126 (f32 out + optional f16 out)
__global__ __launch_bounds__(128)
void add_ln_kernel(const float* __restrict__ x, const float* __restrict__ y,
                   const float* __restrict__ g, const float* __restrict__ bta,
                   float* __restrict__ out, _Float16* __restrict__ outh)
{
  int row = blockIdx.x, tid = threadIdx.x;
  __shared__ float red[128];
  float v = 0.f;
  if (tid < kDM) {
    v = x[(size_t)row * kDM + tid];
    if (y) v += y[(size_t)row * kDM + tid];
  }
  red[tid] = (tid < kDM) ? v : 0.f;
  __syncthreads();
  for (int s = 64; s > 0; s >>= 1) { if (tid < s) red[tid] += red[tid + s]; __syncthreads(); }
  float mean = red[0] * (1.0f / (float)kDM);
  __syncthreads();
  float d = (tid < kDM) ? (v - mean) : 0.f;
  red[tid] = d * d;
  __syncthreads();
  for (int s = 64; s > 0; s >>= 1) { if (tid < s) red[tid] += red[tid + s]; __syncthreads(); }
  float r = rsqrtf(red[0] * (1.0f / (float)kDM) + 1e-5f);
  if (tid < kDM) {
    float o = d * r * g[tid] + bta[tid];
    out[(size_t)row * kDM + tid] = o;
    if (outh) outh[(size_t)row * kDM + tid] = (_Float16)o;
  }
}

// 9. Distilling conv (wrap, k=3) + BN(eval) + ELU
__global__ __launch_bounds__(128)
void convlayer_kernel(const float* __restrict__ x, const float* __restrict__ cl_w,
                      const float* __restrict__ cl_b, const float* __restrict__ bn_g,
                      const float* __restrict__ bn_b, float* __restrict__ out)
{
  int b = blockIdx.x, l = blockIdx.y, c = threadIdx.x;
  __shared__ float rows[3 * kDM];
  #pragma unroll
  for (int it = 0; it < 3; ++it) {
    int e = threadIdx.x + it * 128;
    if (e < 3 * kDM) {
      int kk = e / kDM, i = e - kk * kDM;
      int ll = (l + kk - 1 + kL) & (kL - 1);
      rows[e] = x[((size_t)b * kL + ll) * kDM + i];
    }
  }
  __syncthreads();
  if (c < kDM) {
    float s = cl_b[c];
    const float* w = cl_w + (size_t)c * kDM * 3;
    for (int i = 0; i < kDM; ++i) {
      s += w[i * 3 + 0] * rows[0 * kDM + i];
      s += w[i * 3 + 1] * rows[1 * kDM + i];
      s += w[i * 3 + 2] * rows[2 * kDM + i];
    }
    s = s * (bn_g[c] * rsqrtf(1.0f + 1e-5f)) + bn_b[c];
    s = (s > 0.f) ? s : (__expf(s) - 1.0f);
    out[((size_t)b * kL + l) * kDM + c] = s;
  }
}

// 10. Maxpool k=3 s=2 pad=1 over length (f32 + f16 outputs)
__global__ __launch_bounds__(128)
void maxpool_kernel(const float* __restrict__ x, float* __restrict__ out,
                    _Float16* __restrict__ outh)
{
  int b = blockIdx.x, t = blockIdx.y, c = threadIdx.x;
  if (c >= kDM) return;
  float m = -3.0e38f;
  #pragma unroll
  for (int d = -1; d <= 1; ++d) {
    int l = 2 * t + d;
    if (l >= 0 && l < kL) m = fmaxf(m, x[((size_t)b * kL + l) * kDM + c]);
  }
  size_t idx = ((size_t)b * 64 + t) * kDM + c;
  out[idx] = m;
  outh[idx] = (_Float16)m;
}

// 11. Final head
__global__ __launch_bounds__(128)
void final_kernel(const float* __restrict__ x, const float* __restrict__ fc_w,
                  const float* __restrict__ fc_b, float* __restrict__ out)
{
  int b = blockIdx.x, c = threadIdx.x;
  if (c >= kDM) return;
  float s = fc_b[0];
  for (int t = 0; t < 64; ++t) s += x[((size_t)b * 64 + t) * kDM + c] * fc_w[t];
  out[(size_t)b * kDM + c] = s;
}

// ---------------------------------------------------------------------------
// Host-side helpers
// ---------------------------------------------------------------------------
static inline void launch_gemm(const _Float16* A, const _Float16* Wt, const float* bias,
                               float* C, _Float16* Ch, int M, int N, int K, int act,
                               hipStream_t s)
{
  dim3 g((N + 127) / 128, (M + 63) / 64);
  gemm_kernel<<<g, 256, 0, s>>>(A, Wt, bias, C, Ch, M, N, K, act);
}

static void run_encoder_layer(const float* xin, const _Float16* xinh, int Lq,
                              const _Float16* qkvwt, const float* qkvb,
                              const _Float16* owt, const float* ob,
                              const _Float16* f1wt, const float* f1b,
                              const _Float16* f2wt, const float* f2b,
                              const float* l1g, const float* l1b,
                              const float* l2g, const float* l2b,
                              _Float16* qkvh, _Float16* ctxh,
                              float* tmp, float* xx, _Float16* xxh, _Float16* midh,
                              float* xout, _Float16* xouth, hipStream_t s)
{
  const int M = kB * Lq;
  const float scale = 0.2182178902359924f;  // 1/sqrt(21)
  launch_gemm(xinh, qkvwt, qkvb, nullptr, qkvh, M, kQKV, kDM, 0, s);
  if (Lq == 128)
    attention_kernel<128><<<dim3(kNH, kB), 256, 0, s>>>(qkvh, ctxh, scale);
  else
    attention_kernel<64><<<dim3(kNH, kB), 256, 0, s>>>(qkvh, ctxh, scale);
  launch_gemm(ctxh, owt, ob, tmp, nullptr, M, kDM, kDM, 0, s);
  add_ln_kernel<<<M, 128, 0, s>>>(xin, tmp, l1g, l1b, xx, xxh);
  launch_gemm(xxh, f1wt, f1b, nullptr, midh, M, kDFF, kDM, 1, s);
  launch_gemm(midh, f2wt, f2b, tmp, nullptr, M, kDM, kDFF, 0, s);
  add_ln_kernel<<<M, 128, 0, s>>>(xx, tmp, l2g, l2b, xout, xouth);
}

extern "C" void kernel_launch(void* const* d_in, const int* in_sizes, int n_in,
                              void* d_out, int out_size, void* d_ws, size_t ws_size,
                              hipStream_t stream)
{
  (void)in_sizes; (void)n_in; (void)out_size; (void)ws_size;
  const float* x_enc    = (const float*)d_in[0];
  const float* x_mark   = (const float*)d_in[1];
  const float* token_w  = (const float*)d_in[2];
  const float* time_w   = (const float*)d_in[3];
  const float* time_b   = (const float*)d_in[4];
  const float* od_fc_w  = (const float*)d_in[5];
  const float* od_bn_g  = (const float*)d_in[6];
  const float* od_bn_b  = (const float*)d_in[7];
  const float* od_ch_w  = (const float*)d_in[8];
  const float* od_ch_b  = (const float*)d_in[9];
  const float* od_fil_w = (const float*)d_in[10];
  const float* od_fil_b = (const float*)d_in[11];
  const float* od_sp_w  = (const float*)d_in[12];
  const float* od_sp_b  = (const float*)d_in[13];
  const float* od_ker_w = (const float*)d_in[14];
  const float* od_ker_b = (const float*)d_in[15];
  const float* od_weight= (const float*)d_in[16];
  const float* q_w  = (const float*)d_in[17];
  const float* q_b  = (const float*)d_in[18];
  const float* k_w  = (const float*)d_in[19];
  const float* k_b  = (const float*)d_in[20];
  const float* v_w  = (const float*)d_in[21];
  const float* v_b  = (const float*)d_in[22];
  const float* o_w  = (const float*)d_in[23];
  const float* o_b  = (const float*)d_in[24];
  const float* f1_w = (const float*)d_in[25];
  const float* f1_b = (const float*)d_in[26];
  const float* f2_w = (const float*)d_in[27];
  const float* f2_b = (const float*)d_in[28];
  const float* l1_g = (const float*)d_in[29];
  const float* l1_b = (const float*)d_in[30];
  const float* l2_g = (const float*)d_in[31];
  const float* l2_b = (const float*)d_in[32];
  const float* cl_w = (const float*)d_in[33];
  const float* cl_b = (const float*)d_in[34];
  const float* cl_bn_g = (const float*)d_in[35];
  const float* cl_bn_b = (const float*)d_in[36];
  const float* eln_g = (const float*)d_in[37];
  const float* eln_b = (const float*)d_in[38];
  const float* fc_w  = (const float*)d_in[39];
  const float* fc_b  = (const float*)d_in[40];
  float* out = (float*)d_out;

  // Workspace layout
  float* W = (float*)d_ws;
  size_t off = 0;
  auto allocF = [&](size_t n) { float* p = W + off; off += n; return p; };
  auto allocH = [&](size_t n) { _Float16* p = (_Float16*)(W + off); off += (n + 1) / 2; return p; };
  const size_t SZ1 = (size_t)kB * kL * kDM;
  const size_t SZ2 = (size_t)kB * 64 * kDM;
  float*    emb   = allocF((size_t)kB * kL * kDE);
  float*    gap   = allocF((size_t)kB * kDE);
  float*    attv  = allocF((size_t)kB * 264);
  _Float16* xch   = allocH((size_t)kB * kL * kDE);
  _Float16* aggh  = allocH((size_t)kB * kDE * 384);
  float*    x1    = allocF(SZ1);
  _Float16* x1h   = allocH(SZ1);
  _Float16* qkvh  = allocH((size_t)kB * kL * kQKV);
  _Float16* ctxh  = allocH(SZ1);
  float*    tmp   = allocF(SZ1);
  float*    xx    = allocF(SZ1);
  _Float16* xxh   = allocH(SZ1);
  _Float16* midh  = allocH((size_t)kB * kL * kDFF);
  float*    xo1   = allocF(SZ1);
  float*    cnv   = allocF(SZ1);
  float*    pool  = allocF(SZ2);
  _Float16* poolh = allocH(SZ2);
  float*    xo2   = allocF(SZ2);
  float*    xf    = allocF(SZ2);
  // f16 transposed weights
  _Float16* qkvwt = allocH((size_t)2 * kQKV * kDM);   // (378 x 126) per layer
  _Float16* owt   = allocH((size_t)2 * kDM * kDM);
  _Float16* f1wt  = allocH((size_t)2 * kDFF * kDM);
  _Float16* f2wt  = allocH((size_t)2 * kDM * kDFF);
  float*    qkvb  = allocF((size_t)2 * kQKV);

  // 0: weight repack
  {
    dim3 gP((kDM + 31) / 32, (kDM + 31) / 32);
    dim3 gF1((kDM + 31) / 32, (kDFF + 31) / 32);
    dim3 gF2((kDFF + 31) / 32, (kDM + 31) / 32);
    for (int li = 0; li < 2; ++li) {
      _Float16* qkvwt_l = qkvwt + (size_t)li * kQKV * kDM;
      transpose_f16_kernel<<<gP, 256, 0, stream>>>(q_w + li*kDM*kDM, qkvwt_l,                        kDM, kDM);
      transpose_f16_kernel<<<gP, 256, 0, stream>>>(k_w + li*kDM*kDM, qkvwt_l + (size_t)kDM*kDM,      kDM, kDM);
      transpose_f16_kernel<<<gP, 256, 0, stream>>>(v_w + li*kDM*kDM, qkvwt_l + (size_t)2*kDM*kDM,    kDM, kDM);
      transpose_f16_kernel<<<gP, 256, 0, stream>>>(o_w + li*kDM*kDM, owt + (size_t)li*kDM*kDM,       kDM, kDM);
      transpose_f16_kernel<<<gF1, 256, 0, stream>>>(f1_w + (size_t)li*kDM*kDFF, f1wt + (size_t)li*kDFF*kDM, kDM, kDFF);
      transpose_f16_kernel<<<gF2, 256, 0, stream>>>(f2_w + (size_t)li*kDFF*kDM, f2wt + (size_t)li*kDM*kDFF, kDFF, kDM);
    }
    qkv_bias_kernel<<<2, 128, 0, stream>>>(q_b, k_b, v_b, qkvb);
  }

  // 1-4: embedding + ODConv machinery
  embed_kernel<<<dim3(kB, kL), 128, 0, stream>>>(x_enc, x_mark, token_w, time_w, time_b, emb);
  gap_kernel<<<kB, 128, 0, stream>>>(emb, gap);
  odatt_kernel<<<kB, 128, 0, stream>>>(gap, od_fc_w, od_bn_g, od_bn_b, od_ch_w, od_ch_b,
                                       od_fil_w, od_fil_b, od_sp_w, od_sp_b,
                                       od_ker_w, od_ker_b, attv);
  xch_kernel<<<dim3(kB, kL), 128, 0, stream>>>(emb, attv, xch);
  aggw_kernel<<<dim3(kB, kDE), 128, 0, stream>>>(od_weight, attv, aggh);

  // 5: dynamic conv (double-buffered TDM) -> x1 (f32+f16)
  dynconv_kernel<<<dim3(2, kB), 256, 0, stream>>>(aggh, xch, attv, x1, x1h);

  // 6: encoder layer 0 (Lq=128)
  run_encoder_layer(x1, x1h, 128,
                    qkvwt, qkvb, owt, o_b,
                    f1wt, f1_b, f2wt, f2_b, l1_g, l1_b, l2_g, l2_b,
                    qkvh, ctxh, tmp, xx, xxh, midh, xo1, nullptr, stream);

  // 7: distilling conv + ELU + maxpool
  convlayer_kernel<<<dim3(kB, kL), 128, 0, stream>>>(xo1, cl_w, cl_b, cl_bn_g, cl_bn_b, cnv);
  maxpool_kernel<<<dim3(kB, 64), 128, 0, stream>>>(cnv, pool, poolh);

  // 8: encoder layer 1 (Lq=64)
  run_encoder_layer(pool, poolh, 64,
                    qkvwt + (size_t)kQKV*kDM, qkvb + kQKV, owt + (size_t)kDM*kDM, o_b + kDM,
                    f1wt + (size_t)kDFF*kDM, f1_b + kDFF, f2wt + (size_t)kDM*kDFF, f2_b + kDM,
                    l1_g + kDM, l1_b + kDM, l2_g + kDM, l2_b + kDM,
                    qkvh, ctxh, tmp, xx, xxh, midh, xo2, nullptr, stream);

  // 9: final LayerNorm + projection head
  add_ln_kernel<<<kB * 64, 128, 0, stream>>>(xo2, nullptr, eln_g, eln_b, xf, nullptr);
  final_kernel<<<kB, 128, 0, stream>>>(xf, fc_w, fc_b, out);
}
